// LocalGlobalSemantics_82703890252358
// MI455X (gfx1250) — matrix operand order repaired and checked by reading, not compiled, not run on hardware
//
#include <hip/hip_runtime.h>
#include <hip/hip_bf16.h>

// ---------------------------------------------------------------------------
// CDNA5 (gfx1250, wave32) linear-attention block.
// Heavy matmuls -> v_wmma_f32_16x16x32_bf16 with fragment-packed weights
// (B frag = one 32B vector load / lane, A frag = two 16B loads / lane).
// ---------------------------------------------------------------------------

typedef __bf16 bf16;
typedef __attribute__((ext_vector_type(16))) __bf16 v16bf;
typedef __attribute__((ext_vector_type(8)))  __bf16 v8bf;
typedef __attribute__((ext_vector_type(8)))  float  v8f;
typedef __attribute__((ext_vector_type(4)))  int    v4i;

#define NB     16
#define CCH    512
#define HWPX   4096
#define INTER  512
#define HEADS  64
#define HDIM   8
#define GROUPS 8
#define ICG    64
#define OCG    64
#define KKC    3136        // 64*49, conv im2col reduction
#define KTC    98          // KKC/32 k-tiles for conv

#define LG_WMMA(a, b, c) __builtin_amdgcn_wmma_f32_16x16x32_bf16( \
    false, (a), false, (b), (short)0, (c), false, false)

__device__ __forceinline__ v16bf lg_cat8(v8bf lo, v8bf hi) {
  return __builtin_shufflevector(lo, hi, 0, 1, 2, 3, 4, 5, 6, 7,
                                 8, 9, 10, 11, 12, 13, 14, 15);
}

// ======================= prep kernels ======================================

__global__ void lg_cast_x_bf16(const float* __restrict__ x, bf16* __restrict__ xb, int n) {
  int i = blockIdx.x * 256 + threadIdx.x;
  if (i < n) xb[i] = (bf16)x[i];
}

// Fragment-major packing for WMMA B matrices (16x16x32 bf16):
//   element (k, n) -> flat = (ktile*NTILES + ntile)*512 + lane*16 + e
//   lane = ((k>>3)&1)*16 + (n&15),  e = (k&7) | ((k>>1)&8),  ktile = k>>5, ntile = n>>4

// conv_w [512][64][7][7] -> per-group frag-packed [g][kt=0..97][nt=0..3][lane][e]
__global__ void lg_pack_conv_w(const float* __restrict__ w, bf16* __restrict__ wf) {
  int i = blockIdx.x * 256 + threadIdx.x;
  if (i >= 512 * KKC) return;
  int ocg  = i / KKC;            // global out channel = n of GEMM
  int kidx = i - ocg * KKC;      // k of GEMM
  int g   = ocg >> 6;
  int oc  = ocg & 63;
  int kt  = kidx >> 5;
  int hi  = (kidx >> 3) & 1;
  int e   = (kidx & 7) | ((kidx >> 1) & 8);
  int ntl = oc >> 4;
  int l16 = oc & 15;
  size_t dst = ((((size_t)g * KTC + kt) * 4 + ntl) * 32 + hi * 16 + l16) * 16 + e;
  wf[dst] = (bf16)w[i];
}

// W [n][k] (torch Linear) -> frag-packed [kt=0..15][ntl=0..31][lane][e]
__global__ void lg_pack_wt_frag(const float* __restrict__ w, bf16* __restrict__ wf) {
  int i = blockIdx.x * 256 + threadIdx.x;
  if (i >= 512 * 512) return;
  int n = i >> 9, k = i & 511;
  int kt  = k >> 5;
  int hi  = (k >> 3) & 1;
  int e   = (k & 7) | ((k >> 1) & 8);
  int ntl = n >> 4;
  int l16 = n & 15;
  size_t dst = (((size_t)kt * 32 + ntl) * 32 + hi * 16 + l16) * 16 + e;
  wf[dst] = (bf16)w[i];
}

// ======================= grouped conv as implicit GEMM =====================
// grid (HW/128, GROUPS, NB), block 256 (8 waves); wave tile: 16 px x 64 oc.

__device__ __forceinline__ bf16 lg_im2col(const bf16* __restrict__ xg,
                                          int py, int px, int k) {
  int ic = k / 49;
  int r  = k - ic * 49;
  int ky = r / 7;
  int kx = r - ky * 7;
  int y = py + ky - 3, x = px + kx - 3;
  if ((unsigned)y > 63u || (unsigned)x > 63u) return (bf16)0.0f;
  return xg[(size_t)ic * HWPX + (y << 6) + x];
}

__global__ void lg_conv_wmma(const bf16* __restrict__ xb,
                             const bf16* __restrict__ wf,
                             const float* __restrict__ conv_b,
                             float* __restrict__ xc) {
  const int tp   = blockIdx.x;
  const int g    = blockIdx.y;
  const int n    = blockIdx.z;
  const int tid  = threadIdx.x;
  const int wave = tid >> 5;
  const int lane = tid & 31;
  const int l16  = lane & 15;
  const int hi   = lane >> 4;

  const bf16* wg = wf + (size_t)g * KTC * 4 * 512;
  const bf16* xg = xb + ((size_t)n * CCH + g * ICG) * HWPX;

  const int p  = tp * 128 + wave * 16 + l16;   // this lane's A-frag pixel row
  const int py = p >> 6;
  const int px = p & 63;

  v8f acc[4] = {};

  for (int k0 = 0, kt = 0; k0 < KKC; k0 += 32, ++kt) {
    v16bf a;
#pragma unroll
    for (int e = 0; e < 8; ++e) {
      a[e]     = lg_im2col(xg, py, px, k0 + hi * 8 + e);
      a[8 + e] = lg_im2col(xg, py, px, k0 + hi * 8 + 16 + e);
    }
    const bf16* bp = wg + (size_t)kt * 4 * 512 + lane * 16;
#pragma unroll
    for (int nt = 0; nt < 4; ++nt) {
      v16bf b = *(const v16bf*)(bp + nt * 512);
      acc[nt] = LG_WMMA(a, b, acc[nt]);
    }
  }

#pragma unroll
  for (int nt = 0; nt < 4; ++nt) {
    int   oc   = nt * 16 + l16;
    float bias = conv_b[g * OCG + oc];
#pragma unroll
    for (int j = 0; j < 8; ++j) {
      int pr = tp * 128 + wave * 16 + j + 8 * hi;
      xc[((size_t)n * HWPX + pr) * INTER + g * OCG + oc] = acc[nt][j] + bias;
    }
  }
}

// ======================= LayerNorm (wave32 shuffle reduce) =================
__global__ void lg_layernorm(const float* __restrict__ xc,
                             const float* __restrict__ gam,
                             const float* __restrict__ bet,
                             bf16* __restrict__ xn) {
  int row  = blockIdx.x * 8 + (threadIdx.x >> 5);
  int lane = threadIdx.x & 31;
  const float* src = xc + (size_t)row * INTER + lane * 16;

  float vals[16];
  float s = 0.f;
#pragma unroll
  for (int i = 0; i < 16; ++i) { vals[i] = src[i]; s += vals[i]; }
#pragma unroll
  for (int off = 16; off; off >>= 1) s += __shfl_xor(s, off, 32);
  float mu = s * (1.f / 512.f);

  float s2 = 0.f;
#pragma unroll
  for (int i = 0; i < 16; ++i) { float d = vals[i] - mu; s2 += d * d; }
#pragma unroll
  for (int off = 16; off; off >>= 1) s2 += __shfl_xor(s2, off, 32);
  float rstd = rsqrtf(s2 * (1.f / 512.f) + 1e-5f);

  bf16* dst = xn + (size_t)row * INTER + lane * 16;
#pragma unroll
  for (int i = 0; i < 16; ++i) {
    int c = lane * 16 + i;
    dst[i] = (bf16)((vals[i] - mu) * rstd * gam[c] + bet[c]);
  }
}

// ======================= fused Q/K/V projections (WMMA) ====================
// One A-fragment feeds all three projections: 12 v_wmma per K-step.
// grid (M/128, 512/64), block 256.
__global__ void lg_gemm_qkv(const bf16* __restrict__ xn,
                            const bf16* __restrict__ wqf,
                            const bf16* __restrict__ wkf,
                            const bf16* __restrict__ wvf,
                            const float* __restrict__ qbias,
                            const float* __restrict__ kbias,
                            const float* __restrict__ vbias,
                            bf16* __restrict__ qout,
                            float* __restrict__ kTout,
                            bf16* __restrict__ vout) {
  const int m0   = blockIdx.x * 128;
  const int n0   = blockIdx.y * 64;
  const int tid  = threadIdx.x;
  const int wave = tid >> 5;
  const int lane = tid & 31;
  const int l16  = lane & 15;
  const int hi   = lane >> 4;

  const int row = m0 + wave * 16 + l16;            // A-frag row of this lane
  const bf16* arow = xn + (size_t)row * 512 + hi * 8;

  v8f acc[3][4] = {};

  for (int k0 = 0, kt = 0; k0 < 512; k0 += 32, ++kt) {
    v8bf alo = *(const v8bf*)(arow + k0);
    v8bf ahi = *(const v8bf*)(arow + k0 + 16);
    v16bf a = lg_cat8(alo, ahi);
    if (k0 + 32 < 512) __builtin_prefetch(arow + k0 + 32, 0, 1);

    const size_t boff = ((size_t)kt * 32 + (n0 >> 4)) * 512 + lane * 16;
#pragma unroll
    for (int w3 = 0; w3 < 3; ++w3) {
      const bf16* bp = (w3 == 0 ? wqf : (w3 == 1 ? wkf : wvf)) + boff;
#pragma unroll
      for (int nt = 0; nt < 4; ++nt) {
        v16bf b = *(const v16bf*)(bp + nt * 512);
        acc[w3][nt] = LG_WMMA(a, b, acc[w3][nt]);
      }
    }
  }

#pragma unroll
  for (int w3 = 0; w3 < 3; ++w3) {
    const float* bias = (w3 == 0 ? qbias : (w3 == 1 ? kbias : vbias));
#pragma unroll
    for (int nt = 0; nt < 4; ++nt) {
      int col = n0 + nt * 16 + l16;
      float bv = bias[col];
#pragma unroll
      for (int j = 0; j < 8; ++j) {
        int   r   = m0 + wave * 16 + j + 8 * hi;
        float val = acc[w3][nt][j] + bv;
        if (w3 == 0) {
          qout[(size_t)r * 512 + col] = (bf16)val;
        } else if (w3 == 1) {
          int nb = r >> 12, hw = r & 4095;
          kTout[((size_t)nb * 512 + col) * HWPX + hw] = val;   // transposed
        } else {
          vout[(size_t)r * 512 + col] = (bf16)val;
        }
      }
    }
  }
}

// ======================= softmax over tokens + corr ========================
__device__ __forceinline__ void lg_wait_async() {
#if __has_builtin(__builtin_amdgcn_s_wait_asynccnt)
  __builtin_amdgcn_s_wait_asynccnt(0);
#else
  asm volatile("s_wait_asynccnt 0x0" ::: "memory");
#endif
}

__global__ void lg_softmax_corr(const float* __restrict__ kT,
                                const bf16* __restrict__ v,
                                float* __restrict__ corr) {
  __shared__ __align__(16) float kcol[HWPX];
  __shared__ float red[256];

  const int chn = blockIdx.x;          // n*512 + ch
  const int n   = chn >> 9;
  const int ch  = chn & 511;
  const int h   = ch >> 3;
  const int d   = ch & 7;
  const int tid = threadIdx.x;

  const float* src = kT + (size_t)chn * HWPX;

  float lmax = -3.0e38f;
#if __has_builtin(__builtin_amdgcn_global_load_async_to_lds_b128)
  // CDNA5 async copy: stream the 16KB k-column into LDS, then reduce from LDS
  for (int i = tid * 4; i < HWPX; i += 1024) {
    __builtin_amdgcn_global_load_async_to_lds_b128(
        (__attribute__((address_space(1))) v4i*)(src + i),
        (__attribute__((address_space(3))) v4i*)&kcol[i], 0, 0);
  }
  lg_wait_async();
  __syncthreads();
  for (int i = tid; i < HWPX; i += 256) lmax = fmaxf(lmax, kcol[i]);
#else
  for (int i = tid; i < HWPX; i += 256) {
    float t = src[i];
    kcol[i] = t;
    lmax = fmaxf(lmax, t);
  }
#endif
  red[tid] = lmax; __syncthreads();
  for (int s = 128; s; s >>= 1) {
    if (tid < s) red[tid] = fmaxf(red[tid], red[tid + s]);
    __syncthreads();
  }
  float mx = red[0]; __syncthreads();

  float lsum = 0.f;
  for (int i = tid; i < HWPX; i += 256) {
    float e = __expf(kcol[i] - mx);
    kcol[i] = e;
    lsum += e;
  }
  red[tid] = lsum; __syncthreads();
  for (int s = 128; s; s >>= 1) {
    if (tid < s) red[tid] += red[tid + s];
    __syncthreads();
  }
  float inv = 1.f / red[0]; __syncthreads();

  float pa[8] = {0.f, 0.f, 0.f, 0.f, 0.f, 0.f, 0.f, 0.f};
  const bf16* vb = v + (size_t)n * HWPX * 512 + h * 8;
  for (int i = tid; i < HWPX; i += 256) {
    float w = kcol[i];
    const bf16* vr = vb + (size_t)i * 512;
#pragma unroll
    for (int e = 0; e < 8; ++e) pa[e] += w * (float)vr[e];
  }
#pragma unroll
  for (int e = 0; e < 8; ++e) {
    red[tid] = pa[e]; __syncthreads();
    for (int s = 128; s; s >>= 1) {
      if (tid < s) red[tid] += red[tid + s];
      __syncthreads();
    }
    if (tid == 0) corr[(((size_t)n * HEADS + h) * HDIM + d) * HDIM + e] = red[0] * inv;
    __syncthreads();
  }
}

// ======================= atten = q @ corr (tiny per-token) =================
__global__ void lg_atten(const bf16* __restrict__ q,
                         const float* __restrict__ corr,
                         bf16* __restrict__ atten) {
  int token = blockIdx.x * 8 + (threadIdx.x >> 5);
  int lane  = threadIdx.x & 31;
  int n     = token >> 12;
#pragma unroll
  for (int hh = 0; hh < 2; ++hh) {
    int h = lane * 2 + hh;
    const bf16*  qp = q    + (size_t)token * 512 + h * 8;
    const float* cp = corr + ((size_t)n * HEADS + h) * 64;
    float qv[8];
#pragma unroll
    for (int dd = 0; dd < 8; ++dd) qv[dd] = (float)qp[dd];
    bf16* op = atten + (size_t)token * 512 + h * 8;
#pragma unroll
    for (int e = 0; e < 8; ++e) {
      float s = 0.f;
#pragma unroll
      for (int dd = 0; dd < 8; ++dd) s += qv[dd] * cp[dd * 8 + e];
      op[e] = (bf16)s;
    }
  }
}

// ======================= output projection + residual ======================
__global__ void lg_gemm_out(const bf16* __restrict__ atten,
                            const bf16* __restrict__ wpf,
                            const float* __restrict__ pb,
                            const float* __restrict__ beta,
                            const float* __restrict__ x,
                            float* __restrict__ out) {
  const int m0   = blockIdx.x * 128;
  const int n0   = blockIdx.y * 64;
  const int tid  = threadIdx.x;
  const int wave = tid >> 5;
  const int lane = tid & 31;
  const int l16  = lane & 15;
  const int hi   = lane >> 4;

  const int row = m0 + wave * 16 + l16;
  const bf16* arow = atten + (size_t)row * 512 + hi * 8;

  v8f acc[4] = {};

  for (int k0 = 0, kt = 0; k0 < 512; k0 += 32, ++kt) {
    v8bf alo = *(const v8bf*)(arow + k0);
    v8bf ahi = *(const v8bf*)(arow + k0 + 16);
    v16bf a = lg_cat8(alo, ahi);
    if (k0 + 32 < 512) __builtin_prefetch(arow + k0 + 32, 0, 1);

    const bf16* bp = wpf + ((size_t)kt * 32 + (n0 >> 4)) * 512 + lane * 16;
#pragma unroll
    for (int nt = 0; nt < 4; ++nt) {
      v16bf b = *(const v16bf*)(bp + nt * 512);
      acc[nt] = LG_WMMA(a, b, acc[nt]);
    }
  }

  const float bt = beta[0];
#pragma unroll
  for (int nt = 0; nt < 4; ++nt) {
    int   col = n0 + nt * 16 + l16;
    float bv  = pb[col];
#pragma unroll
    for (int j = 0; j < 8; ++j) {
      int    r  = m0 + wave * 16 + j + 8 * hi;
      int    nb = r >> 12, hw = r & 4095;
      size_t oi = ((size_t)nb * CCH + col) * HWPX + hw;
      out[oi] = bt * (acc[nt][j] + bv) + x[oi];
    }
  }
}

// ======================= host-side launch ==================================

extern "C" void kernel_launch(void* const* d_in, const int* in_sizes, int n_in,
                              void* d_out, int out_size, void* d_ws, size_t ws_size,
                              hipStream_t stream) {
  (void)in_sizes; (void)n_in; (void)out_size; (void)ws_size;

  const float* x      = (const float*)d_in[0];
  const float* conv_w = (const float*)d_in[1];
  const float* conv_b = (const float*)d_in[2];
  const float* ln_g   = (const float*)d_in[3];
  const float* ln_b   = (const float*)d_in[4];
  const float* qw     = (const float*)d_in[5];
  const float* qb     = (const float*)d_in[6];
  const float* kw     = (const float*)d_in[7];
  const float* kb     = (const float*)d_in[8];
  const float* vw     = (const float*)d_in[9];
  const float* vb     = (const float*)d_in[10];
  const float* pw     = (const float*)d_in[11];
  const float* pb     = (const float*)d_in[12];
  const float* beta   = (const float*)d_in[13];
  float*       out    = (float*)d_out;

  size_t off = 0;
  auto carve = [&](size_t bytes) -> void* {
    void* p = (char*)d_ws + off;
    off += (bytes + 255) & ~(size_t)255;
    return p;
  };
  const size_t NE = (size_t)NB * CCH * HWPX;
  bf16*  xbf  = (bf16*) carve(NE * 2);
  bf16*  wcf  = (bf16*) carve((size_t)GROUPS * KTC * 4 * 512 * 2);
  bf16*  wqf  = (bf16*) carve(512 * 512 * 2);
  bf16*  wkf  = (bf16*) carve(512 * 512 * 2);
  bf16*  wvf  = (bf16*) carve(512 * 512 * 2);
  bf16*  wpf  = (bf16*) carve(512 * 512 * 2);
  float* xc   = (float*)carve(NE * 4);
  bf16*  xn   = (bf16*) carve(NE * 2);
  bf16*  qo   = (bf16*) carve(NE * 2);
  float* kT   = (float*)carve(NE * 4);
  bf16*  vo   = (bf16*) carve(NE * 2);
  float* corr = (float*)carve((size_t)NB * HEADS * HDIM * HDIM * 4);
  bf16*  attn = (bf16*) carve(NE * 2);

  lg_cast_x_bf16<<<(int)(NE / 256), 256, 0, stream>>>(x, xbf, (int)NE);
  lg_pack_conv_w<<<(512 * KKC + 255) / 256, 256, 0, stream>>>(conv_w, wcf);
  lg_pack_wt_frag<<<1024, 256, 0, stream>>>(qw, wqf);
  lg_pack_wt_frag<<<1024, 256, 0, stream>>>(kw, wkf);
  lg_pack_wt_frag<<<1024, 256, 0, stream>>>(vw, wvf);
  lg_pack_wt_frag<<<1024, 256, 0, stream>>>(pw, wpf);

  lg_conv_wmma<<<dim3(HWPX / 128, GROUPS, NB), 256, 0, stream>>>(xbf, wcf, conv_b, xc);

  lg_layernorm<<<(NB * HWPX) / 8, 256, 0, stream>>>(xc, ln_g, ln_b, xn);

  lg_gemm_qkv<<<dim3((NB * HWPX) / 128, INTER / 64), 256, 0, stream>>>(
      xn, wqf, wkf, wvf, qb, kb, vb, qo, kT, vo);

  lg_softmax_corr<<<NB * INTER, 256, 0, stream>>>(kT, vo, corr);

  lg_atten<<<(NB * HWPX) / 8, 256, 0, stream>>>(qo, corr, attn);

  lg_gemm_out<<<dim3((NB * HWPX) / 128, CCH / 64), 256, 0, stream>>>(
      attn, wpf, pb, beta, x, out);
}
